// SparseMSDeformableAttention_10651518894417
// MI455X (gfx1250) — compile-verified, matrix operand order checked by reading, not run on hardware
//
#include <hip/hip_runtime.h>
#include <hip/hip_bf16.h>
#include <math.h>

typedef __attribute__((ext_vector_type(16))) __bf16 v16bf;
typedef __attribute__((ext_vector_type(8)))  float  v8f;

// ---------------------------------------------------------------------------
// Weight prepack: W[K,Ncols] f32 -> bf16 in WMMA B-fragment order.
// For col-tile ct, k-step ks, lane: 16 contiguous bf16 (32B) holding
//   frag[i]   = W[ks*32 + kb + i,      ct*16 + (lane&15)]
//   frag[8+i] = W[ks*32 + 16 + kb + i, ct*16 + (lane&15)]   (kb = (lane>>4)*8)
// so the GEMM inner loop reads one aligned v16bf per (tile, k-step).
// ---------------------------------------------------------------------------
__global__ void pack_wfrag_kernel(const float* __restrict__ W,
                                  __bf16* __restrict__ Wp,
                                  int Ncols, int K) {
  const int lane  = threadIdx.x & 31;
  const int wave  = threadIdx.x >> 5;
  const int Kd32  = K >> 5;
  const int pairs = (Ncols >> 4) * Kd32;
  const int id    = blockIdx.x * 8 + wave;
  if (id >= pairs) return;
  const int ct = id / Kd32;
  const int ks = id % Kd32;
  const int kb  = (lane >> 4) * 8;
  const int col = ct * 16 + (lane & 15);
  __bf16* dst = Wp + ((size_t)id * 32 + lane) * 16;
#pragma unroll
  for (int i = 0; i < 8; ++i) {
    dst[i]     = (__bf16)W[(size_t)(ks * 32 + kb + i) * Ncols + col];
    dst[8 + i] = (__bf16)W[(size_t)(ks * 32 + 16 + kb + i) * Ncols + col];
  }
}

// Load one 16-bf16 A fragment (row `row`, k-base k0+kb) as 4x b128 + cvt_pk.
__device__ __forceinline__ v16bf load_a_frag(const float* __restrict__ A,
                                             size_t rowOff, int k) {
  const float* arow = A + rowOff + k;
  const float4 x0 = ((const float4*)arow)[0];
  const float4 x1 = ((const float4*)arow)[1];
  const float4 x2 = ((const float4*)(arow + 16))[0];
  const float4 x3 = ((const float4*)(arow + 16))[1];
  v16bf a;
  a[0]  = (__bf16)x0.x; a[1]  = (__bf16)x0.y; a[2]  = (__bf16)x0.z; a[3]  = (__bf16)x0.w;
  a[4]  = (__bf16)x1.x; a[5]  = (__bf16)x1.y; a[6]  = (__bf16)x1.z; a[7]  = (__bf16)x1.w;
  a[8]  = (__bf16)x2.x; a[9]  = (__bf16)x2.y; a[10] = (__bf16)x2.z; a[11] = (__bf16)x2.w;
  a[12] = (__bf16)x3.x; a[13] = (__bf16)x3.y; a[14] = (__bf16)x3.z; a[15] = (__bf16)x3.w;
  return a;
}

// ---------------------------------------------------------------------------
// GEMM:  out[M,Ncols] = A[M,K] * W[K,Ncols] + bias[Ncols]
// A is f32 (b128 loads, inline cvt to bf16); W is prepacked bf16 fragments.
// One wave computes a 32-row x 64-col strip: 2 row tiles x 4 col tiles,
// 8 WMMAs per 32-wide k-step with all fragment loads hoisted ahead of the
// WMMA group (one load clause -> one wait -> 8 back-to-back v_wmma).
// 256-thread block = 8 waves = 256 rows; all M here are multiples of 256.
//
// Fragment layouts per CDNA5 ISA 7.12.2 (wave32):
//   A (16x32 bf16): lane<16 -> row=lane,  K = {kb..kb+7, 16+kb..23+kb}, kb=0
//                   lane>=16 -> row=lane-16, same with kb=8
//   C/D (16x16 f32): vgpr v -> row v + 8*(lane>>4), col = lane&15
// ---------------------------------------------------------------------------
template <bool BF16_OUT>
__global__ void gemm_bias_kernel(const float* __restrict__ A,
                                 const __bf16* __restrict__ Wp,
                                 const float* __restrict__ bias,
                                 void* __restrict__ outv,
                                 int M, int Ncols, int K) {
  const int lane    = threadIdx.x & 31;
  const int wave    = threadIdx.x >> 5;
  const int rowBase = (blockIdx.x * 8 + wave) * 32;
  if (rowBase >= M) return;                   // wave-uniform: EXEC stays full

  const int half    = lane >> 4;
  const int l16     = lane & 15;
  const int colBase = blockIdx.y * 64;
  const int kb      = half * 8;
  const int Kd32    = K >> 5;
  const int ct0     = blockIdx.y * 4;

  const size_t rowOff0 = (size_t)(rowBase + l16) * K;
  const size_t rowOff1 = (size_t)(rowBase + 16 + l16) * K;

  v8f acc[2][4] = {};

  for (int k0 = 0; k0 < K; k0 += 32) {
    const int ks = k0 >> 5;
    // All fragment loads up front: 8x b128 (A) + 8x b128 (B)
    v16bf b[4];
#pragma unroll
    for (int t = 0; t < 4; ++t)
      b[t] = *(const v16bf*)(Wp + ((size_t)((ct0 + t) * Kd32 + ks) * 32 + lane) * 16);
    const v16bf a0 = load_a_frag(A, rowOff0, k0 + kb);
    const v16bf a1 = load_a_frag(A, rowOff1, k0 + kb);
    // 8 back-to-back WMMAs on independent accumulators
#pragma unroll
    for (int t = 0; t < 4; ++t)
      acc[0][t] = __builtin_amdgcn_wmma_f32_16x16x32_bf16(
          false, a0, false, b[t], (short)0, acc[0][t], false, false);
#pragma unroll
    for (int t = 0; t < 4; ++t)
      acc[1][t] = __builtin_amdgcn_wmma_f32_16x16x32_bf16(
          false, a1, false, b[t], (short)0, acc[1][t], false, false);
  }

#pragma unroll
  for (int rt = 0; rt < 2; ++rt) {
#pragma unroll
    for (int t = 0; t < 4; ++t) {
      const int   c  = colBase + t * 16 + l16;
      const float bv = bias[c];
#pragma unroll
      for (int v = 0; v < 8; ++v) {
        const int   r   = rowBase + rt * 16 + v + half * 8;
        const float val = acc[rt][t][v] + bv;
        if constexpr (BF16_OUT)
          ((__bf16*)outv)[(size_t)r * Ncols + c] = (__bf16)val;
        else
          ((float*)outv)[(size_t)r * Ncols + c] = val;
      }
    }
  }
}

// ---------------------------------------------------------------------------
// Softmax over 16 contiguous logits per (query, head) row, in place.
// ---------------------------------------------------------------------------
__global__ void softmax16_kernel(float* __restrict__ att, int rows) {
  const int t = blockIdx.x * blockDim.x + threadIdx.x;
  if (t >= rows) return;
  float* p = att + (size_t)t * 16;
  float m = p[0];
#pragma unroll
  for (int i = 1; i < 16; ++i) m = fmaxf(m, p[i]);
  float e[16];
  float s = 0.f;
#pragma unroll
  for (int i = 0; i < 16; ++i) { e[i] = __expf(p[i] - m); s += e[i]; }
  const float inv = 1.f / s;
#pragma unroll
  for (int i = 0; i < 16; ++i) p[i] = e[i] * inv;
}

// ---------------------------------------------------------------------------
// Sampling + attention aggregation.
// One wave per (query, head); lane = channel d in [0,32).
// Per (level, point): bilinear sample of the projected bf16 value map
// (head slice = 32 contiguous bf16 = 64B coalesced across the wave),
// weighted by softmax attention, accumulated into agg[N, 256] f32.
// ---------------------------------------------------------------------------
__global__ void sample_agg_kernel(const float* __restrict__ ref,
                                  const int* __restrict__ qoff,
                                  const float* __restrict__ off_ws,
                                  const float* __restrict__ attn_ws,
                                  const __bf16* __restrict__ vp0,
                                  const __bf16* __restrict__ vp1,
                                  const __bf16* __restrict__ vp2,
                                  const __bf16* __restrict__ vp3,
                                  float* __restrict__ agg, int N) {
  const int wave = threadIdx.x >> 5;
  const int lane = threadIdx.x & 31;
  const int task = blockIdx.x * 8 + wave;
  if (task >= N * 8) return;
  const int n = task >> 3;
  const int h = task & 7;
  const int b = (n >= qoff[1]) ? 1 : 0;   // searchsorted(right)-1, offsets[0]=0

  const float r0 = ref[(size_t)n * 2 + 0];
  const float r1 = ref[(size_t)n * 2 + 1];

  const __bf16* vps[4] = {vp0, vp1, vp2, vp3};
  const int     HL[4]  = {64, 128, 256, 512};
  const float   SC[4]  = {0.125f, 0.25f, 0.5f, 1.0f};  // level / base(512)

  const float* offp = off_ws  + (size_t)n * 256 + h * 32;   // [L][P][2]
  const float* attp = attn_ws + (size_t)n * 128 + h * 16;   // [L][P]

  float acc = 0.f;
#pragma unroll
  for (int l = 0; l < 4; ++l) {
    const int     hl  = HL[l];
    const float   sc  = SC[l];
    const __bf16* vpl = vps[l] + (size_t)b * hl * hl * 256 + h * 32 + lane;
#pragma unroll
    for (int p = 0; p < 4; ++p) {
      const float oi = offp[(l * 4 + p) * 2 + 0];
      const float oj = offp[(l * 4 + p) * 2 + 1];
      const float w  = attp[l * 4 + p];
      const float si = fmaxf((r0 + oi) * sc - 0.5f, 0.f);
      const float sj = fmaxf((r1 + oj) * sc - 0.5f, 0.f);
      const float fi = floorf(si), fj = floorf(sj);
      const float di = si - fi,    dj = sj - fj;
      const int   i0 = (int)fi,    j0 = (int)fj;
      const int i0c = min(i0,     hl - 1);
      const int i1c = min(i0 + 1, hl - 1);
      const int j0c = min(j0,     hl - 1);
      const int j1c = min(j0 + 1, hl - 1);
      const float v00 = (float)vpl[((size_t)i0c * hl + j0c) * 256];
      const float v01 = (float)vpl[((size_t)i0c * hl + j1c) * 256];
      const float v10 = (float)vpl[((size_t)i1c * hl + j0c) * 256];
      const float v11 = (float)vpl[((size_t)i1c * hl + j1c) * 256];
      const float bil = (1.f - di) * ((1.f - dj) * v00 + dj * v01) +
                        di         * ((1.f - dj) * v10 + dj * v11);
      acc += w * bil;
    }
  }
  agg[(size_t)n * 256 + h * 32 + lane] = acc;
}

// ---------------------------------------------------------------------------
// Host launch.  Input order per setup_inputs():
//  0 query[N,256] 1 query_offsets[2] 2 reference_points[N,2]
//  3..6 value0..3[B,h,w,256] 7 W_off 8 b_off 9 W_attn 10 b_attn
//  11 W_val 12 b_val 13 W_out 14 b_out
// ---------------------------------------------------------------------------
extern "C" void kernel_launch(void* const* d_in, const int* in_sizes, int n_in,
                              void* d_out, int out_size, void* d_ws, size_t ws_size,
                              hipStream_t stream) {
  const float* query  = (const float*)d_in[0];
  const int*   qoff   = (const int*)d_in[1];
  const float* ref    = (const float*)d_in[2];
  const float* val[4] = {(const float*)d_in[3], (const float*)d_in[4],
                         (const float*)d_in[5], (const float*)d_in[6]};
  const float* W_off  = (const float*)d_in[7];
  const float* b_off  = (const float*)d_in[8];
  const float* W_attn = (const float*)d_in[9];
  const float* b_attn = (const float*)d_in[10];
  const float* W_val  = (const float*)d_in[11];
  const float* b_val  = (const float*)d_in[12];
  const float* W_out  = (const float*)d_in[13];
  const float* b_out  = (const float*)d_in[14];

  const int C = 256;
  const int N = in_sizes[0] / C;            // 16384

  // Workspace layout
  char*   wsb     = (char*)d_ws;
  size_t  cur     = 0;
  float*  off_ws  = (float*)(wsb + cur);  cur += (size_t)N * 256 * 4;
  float*  attn_ws = (float*)(wsb + cur);  cur += (size_t)N * 128 * 4;
  float*  agg_ws  = (float*)(wsb + cur);  cur += (size_t)N * 256 * 4;
  __bf16* Wp_off  = (__bf16*)(wsb + cur); cur += (size_t)C * 256 * 2;
  __bf16* Wp_attn = (__bf16*)(wsb + cur); cur += (size_t)C * 128 * 2;
  __bf16* Wp_val  = (__bf16*)(wsb + cur); cur += (size_t)C * 256 * 2;
  __bf16* Wp_out  = (__bf16*)(wsb + cur); cur += (size_t)C * 256 * 2;
  __bf16* vpbase  = (__bf16*)(wsb + cur);
  __bf16* vpl[4];
  size_t  o = 0;
  for (int l = 0; l < 4; ++l) { vpl[l] = vpbase + o; o += (size_t)in_sizes[3 + l]; }

  const dim3 blk(256);

  // 0) prepack weights into WMMA B-fragment bf16 layout
  pack_wfrag_kernel<<<dim3(16), blk, 0, stream>>>(W_off,  Wp_off,  256, C);
  pack_wfrag_kernel<<<dim3(8),  blk, 0, stream>>>(W_attn, Wp_attn, 128, C);
  pack_wfrag_kernel<<<dim3(16), blk, 0, stream>>>(W_val,  Wp_val,  256, C);
  pack_wfrag_kernel<<<dim3(16), blk, 0, stream>>>(W_out,  Wp_out,  256, C);

  // 1) query linear projections (WMMA bf16)
  gemm_bias_kernel<false><<<dim3(N / 256, 256 / 64), blk, 0, stream>>>(
      query, Wp_off, b_off, off_ws, N, 256, C);
  gemm_bias_kernel<false><<<dim3(N / 256, 128 / 64), blk, 0, stream>>>(
      query, Wp_attn, b_attn, attn_ws, N, 128, C);

  // 2) per-head softmax over 16 (level, point) logits
  softmax16_kernel<<<dim3((N * 8 + 255) / 256), blk, 0, stream>>>(attn_ws, N * 8);

  // 3) value projection for all pyramid levels -> bf16 vp (the big WMMA GEMM)
  for (int l = 0; l < 4; ++l) {
    const int Ml = in_sizes[3 + l] / C;     // B*h*w rows (multiple of 256)
    gemm_bias_kernel<true><<<dim3((Ml + 255) / 256, 256 / 64), blk, 0, stream>>>(
        val[l], Wp_val, b_val, vpl[l], Ml, 256, C);
  }

  // 4) bilinear sampling + attention-weighted aggregation
  sample_agg_kernel<<<dim3((N * 8) / 8), blk, 0, stream>>>(
      ref, qoff, off_ws, attn_ws, vpl[0], vpl[1], vpl[2], vpl[3], agg_ws, N);

  // 5) output projection
  gemm_bias_kernel<false><<<dim3(N / 256, 256 / 64), blk, 0, stream>>>(
      agg_ws, Wp_out, b_out, (float*)d_out, N, 256, C);
}
